// PointerNetwork_28003186770230
// MI455X (gfx1250) — compile-verified, matrix operand order checked by reading, not run on hardware
//
#include <hip/hip_runtime.h>
#include <hip/hip_bf16.h>
#include <math.h>

#define TQ   64
#define TP   512
#define BB   64
#define HH   768
#define ATT  75
#define ATTP 80      // ATT padded to 5 WMMA tiles
#define G3H  2304    // 3*H
#define NEGV (-1e30f)

typedef __attribute__((ext_vector_type(16))) __bf16 v16bf;
typedef __attribute__((ext_vector_type(2)))  __bf16 v2bf;
typedef __attribute__((ext_vector_type(8)))  float  v8f;

// Pack two fp32 into two bf16 fragment slots (round-to-nearest-even).
// Prefer the single-instruction v_cvt_pk_bf16_f32 when the builtin exists.
static __device__ __forceinline__ void pk2(v16bf& v, int i, float a, float b) {
#if __has_builtin(__builtin_amdgcn_cvt_pk_bf16_f32)
  v2bf p = __builtin_amdgcn_cvt_pk_bf16_f32(a, b);
#else
  unsigned ua = __float_as_uint(a), ub = __float_as_uint(b);
  ua += 0x7FFFu + ((ua >> 16) & 1u);
  ub += 0x7FFFu + ((ub >> 16) & 1u);
  const unsigned pk = (ua >> 16) | (ub & 0xFFFF0000u);
  v2bf p = __builtin_bit_cast(v2bf, pk);
#endif
  v[i] = p[0];
  v[i + 1] = p[1];
}

// Load one 16-value bf16 WMMA fragment for this lane.
// Per ISA 16-bit A/B layout: VGPRs 0..3 hold K = koff+0..7 (two contiguous
// float4s), VGPRs 4..7 hold K = 16+koff+0..7. `p` = row + k0 + koff (16B aligned).
static __device__ __forceinline__ v16bf load_frag(const float* __restrict__ p) {
  const float4 x0 = *(const float4*)(p);
  const float4 x1 = *(const float4*)(p + 4);
  const float4 y0 = *(const float4*)(p + 16);
  const float4 y1 = *(const float4*)(p + 20);
  v16bf f;
  pk2(f, 0,  x0.x, x0.y); pk2(f, 2,  x0.z, x0.w);
  pk2(f, 4,  x1.x, x1.y); pk2(f, 6,  x1.z, x1.w);
  pk2(f, 8,  y0.x, y0.y); pk2(f, 10, y0.z, y0.w);
  pk2(f, 12, y1.x, y1.y); pk2(f, 14, y1.z, y1.w);
  return f;
}

// C(M x ldc) = A(M x K, row-major, stride lda) @ W(N x K, row-major, stride ldw)^T
// bf16 inputs / f32 accumulate via v_wmma_f32_16x16x32_bf16.
// Each wave computes a 16 x (16*TNC) strip: A fragment loaded once per k-step,
// reused for TNC back-to-back WMMAs. REQUIRES M % 16 == 0 and ldc == tilesN*16.
// W rows with n >= Nvalid are clamped to row Nvalid-1 (pad cols hold garbage
// that consumers never read).
template <int TNC>
__global__ __launch_bounds__(256)
void ptrnet_gemm_wmma(const float* __restrict__ A, int lda,
                      const float* __restrict__ W, int ldw,
                      float* __restrict__ C, int ldc,
                      int Nvalid, int K,
                      int tilesM, int chunksN) {
  const int wave = threadIdx.x >> 5;
  const int lane = threadIdx.x & 31;
  const int wid  = blockIdx.x * (blockDim.x >> 5) + wave;
  if (wid >= tilesM * chunksN) return;
  const int tc  = wid % chunksN;
  const int tm  = wid / chunksN;
  const int tn0 = tc * TNC;

  const int mrow = tm * 16 + (lane & 15);   // A row held by this lane
  const int koff = (lane >> 4) * 8;         // lane-half K offset
  const float* Arow = A + (size_t)mrow * lda + koff;

  const float* Wrow[TNC];
  int ncol[TNC];
#pragma unroll
  for (int j = 0; j < TNC; ++j) {
    ncol[j] = (tn0 + j) * 16 + (lane & 15);
    const int nc = (ncol[j] < Nvalid) ? ncol[j] : (Nvalid - 1);
    Wrow[j] = W + (size_t)nc * ldw + koff;
  }

  v8f acc[TNC];
#pragma unroll
  for (int j = 0; j < TNC; ++j) acc[j] = (v8f){};

  for (int k0 = 0; k0 < K; k0 += 32) {
    if (k0 + 32 < K) __builtin_prefetch(Arow + k0 + 32, 0, 1);
    const v16bf af = load_frag(Arow + k0);
#pragma unroll
    for (int j = 0; j < TNC; ++j) {
      const v16bf bf = load_frag(Wrow[j] + k0);
      acc[j] = __builtin_amdgcn_wmma_f32_16x16x32_bf16(false, af, false, bf,
                                                       (short)0, acc[j],
                                                       false, false);
    }
  }

  // C/D layout: VGPR r -> M = r + (lane>=16 ? 8 : 0), N = lane%16
  const int mbase = tm * 16 + ((lane >> 4) << 3);
#pragma unroll
  for (int j = 0; j < TNC; ++j) {
#pragma unroll
    for (int r = 0; r < 8; ++r) {
      C[(size_t)(mbase + r) * ldc + ncol[j]] = acc[j][r];
    }
  }
}

// Question pooling: logits from Qproj + V_q projection, masked softmax over T,
// weighted sum -> state (B,H). Single block of 256 threads.
__global__ void ptrnet_qpool(const float* __restrict__ Qproj,     // (TQ*B, ATTP)
                             const float* __restrict__ question,  // (TQ,B,H)
                             const float* __restrict__ qmask,     // (TQ,B)
                             const float* __restrict__ Vq,        // (H)
                             const float* __restrict__ Wq1,       // (ATT, 2H)
                             const float* __restrict__ wq2,       // (ATT)
                             float* __restrict__ state) {         // (B,H)
  __shared__ float vqp[ATTP];
  __shared__ float sl[TQ * BB];
  const int tid = threadIdx.x;

  for (int a = tid; a < ATT; a += blockDim.x) {
    const float* wr = Wq1 + (size_t)a * (2 * HH) + HH;
    float acc = 0.f;
    for (int h = 0; h < HH; ++h) acc += Vq[h] * wr[h];
    vqp[a] = acc;
  }
  __syncthreads();

  for (int idx = tid; idx < TQ * BB; idx += blockDim.x) {
    const float* qp = Qproj + (size_t)idx * ATTP;
    float acc = 0.f;
    for (int a = 0; a < ATT; ++a) acc += tanhf(qp[a] + vqp[a]) * wq2[a];
    sl[idx] = (qmask[idx] > 0.f) ? acc : NEGV;
  }
  __syncthreads();

  if (tid < BB) {
    float mx = -INFINITY;
    for (int t = 0; t < TQ; ++t) mx = fmaxf(mx, sl[t * BB + tid]);
    float s = 0.f;
    for (int t = 0; t < TQ; ++t) s += expf(sl[t * BB + tid] - mx);
    const float inv = 1.f / s;
    for (int t = 0; t < TQ; ++t)
      sl[t * BB + tid] = expf(sl[t * BB + tid] - mx) * inv;
  }
  __syncthreads();

  for (int idx = tid; idx < BB * HH; idx += blockDim.x) {
    const int b = idx / HH, h = idx % HH;
    float acc = 0.f;
    for (int t = 0; t < TQ; ++t)
      acc += sl[t * BB + b] * question[((size_t)t * BB + b) * HH + h];
    state[idx] = acc;
  }
}

// Passage attention pass. One block per batch column b. Writes raw logits
// (transposed, (B,TP)) into logits_out; if compute_ci, also writes the
// softmax-weighted sum over passage into cell_input (B,H).
__global__ void ptrnet_pattn(const float* __restrict__ Pproj,    // (TP*B, ATTP)
                             const float* __restrict__ passage,  // (TP,B,H)
                             const float* __restrict__ pmask,    // (TP,B)
                             const float* __restrict__ sproj,    // (B, ATTP)
                             const float* __restrict__ wp2,      // (ATT)
                             float* __restrict__ logits_out,     // (B,TP)
                             float* __restrict__ cell_input,     // (B,H) or null
                             int compute_ci) {
  __shared__ float sp[ATTP];
  __shared__ float sc[TP];
  __shared__ float red[256];
  const int b = blockIdx.x;
  const int tid = threadIdx.x;

  if (tid < ATTP) sp[tid] = sproj[b * ATTP + tid];
  __syncthreads();

  float lmax = -INFINITY;
  for (int t = tid; t < TP; t += blockDim.x) {
    const float* pp = Pproj + ((size_t)t * BB + b) * ATTP;
    float acc = 0.f;
    for (int a = 0; a < ATT; ++a) acc += tanhf(pp[a] + sp[a]) * wp2[a];
    logits_out[(size_t)b * TP + t] = acc;  // raw logits (pre-mask), transposed
    const float ml = (pmask[t * BB + b] > 0.f) ? acc : NEGV;
    sc[t] = ml;
    lmax = fmaxf(lmax, ml);
  }
  red[tid] = lmax;
  __syncthreads();
  for (int s = 128; s > 0; s >>= 1) {
    if (tid < s) red[tid] = fmaxf(red[tid], red[tid + s]);
    __syncthreads();
  }
  const float mx = red[0];
  __syncthreads();

  float lsum = 0.f;
  for (int t = tid; t < TP; t += blockDim.x) {
    const float e = expf(sc[t] - mx);
    sc[t] = e;
    lsum += e;
  }
  red[tid] = lsum;
  __syncthreads();
  for (int s = 128; s > 0; s >>= 1) {
    if (tid < s) red[tid] += red[tid + s];
    __syncthreads();
  }
  const float inv = 1.f / red[0];
  __syncthreads();
  for (int t = tid; t < TP; t += blockDim.x) sc[t] *= inv;
  __syncthreads();

  if (compute_ci) {
    for (int h = tid; h < HH; h += blockDim.x) {
      float acc = 0.f;
      for (int t = 0; t < TP; ++t)
        acc += sc[t] * passage[((size_t)t * BB + b) * HH + h];
      cell_input[(size_t)b * HH + h] = acc;
    }
  }
}

// GRU elementwise gate combine: gi/gh are (B,3H) pre-bias GEMM results.
__global__ void ptrnet_gru(const float* __restrict__ gi,
                           const float* __restrict__ gh,
                           const float* __restrict__ b_ih,
                           const float* __restrict__ b_hh,
                           const float* __restrict__ state,
                           float* __restrict__ state2) {
  const int idx = blockIdx.x * blockDim.x + threadIdx.x;
  if (idx >= BB * HH) return;
  const int b = idx / HH, j = idx % HH;
  const float* gib = gi + (size_t)b * G3H;
  const float* ghb = gh + (size_t)b * G3H;
  const float ir = gib[j]            + b_ih[j];
  const float hr = ghb[j]            + b_hh[j];
  const float iz = gib[HH + j]       + b_ih[HH + j];
  const float hz = ghb[HH + j]       + b_hh[HH + j];
  const float in = gib[2 * HH + j]   + b_ih[2 * HH + j];
  const float hn = ghb[2 * HH + j]   + b_hh[2 * HH + j];
  const float r = 1.f / (1.f + expf(-(ir + hr)));
  const float z = 1.f / (1.f + expf(-(iz + hz)));
  const float n = tanhf(in + r * hn);
  state2[idx] = (1.f - z) * n + z * state[idx];
}

extern "C" void kernel_launch(void* const* d_in, const int* in_sizes, int n_in,
                              void* d_out, int out_size, void* d_ws, size_t ws_size,
                              hipStream_t stream) {
  (void)in_sizes; (void)n_in; (void)out_size; (void)ws_size;
  const float* question = (const float*)d_in[0];
  const float* qmask    = (const float*)d_in[1];
  const float* passage  = (const float*)d_in[2];
  const float* pmask    = (const float*)d_in[3];
  const float* Vq       = (const float*)d_in[4];
  const float* Wq1      = (const float*)d_in[5];
  const float* wq2      = (const float*)d_in[6];
  const float* Wp1      = (const float*)d_in[7];
  const float* wp2      = (const float*)d_in[8];
  const float* W_ih     = (const float*)d_in[9];
  const float* W_hh     = (const float*)d_in[10];
  const float* b_ih     = (const float*)d_in[11];
  const float* b_hh     = (const float*)d_in[12];
  float* out = (float*)d_out;   // [start_logits (B,TP) | end_logits (B,TP)]

  // Workspace layout (floats)
  float* ws     = (float*)d_ws;
  float* Qproj  = ws;                              // (TQ*B, ATTP)
  float* Pproj  = Qproj  + (size_t)TQ * BB * ATTP; // (TP*B, ATTP)
  float* state  = Pproj  + (size_t)TP * BB * ATTP; // (B,H)
  float* sproj1 = state  + (size_t)BB * HH;        // (B,ATTP)
  float* cellin = sproj1 + (size_t)BB * ATTP;      // (B,H)
  float* gi     = cellin + (size_t)BB * HH;        // (B,3H)
  float* gh     = gi     + (size_t)BB * G3H;       // (B,3H)
  float* state2 = gh     + (size_t)BB * G3H;       // (B,H)
  float* sproj2 = state2 + (size_t)BB * HH;        // (B,ATTP)

  const dim3 blk(256);
  const int wpb = 256 / 32;  // waves per block

  // 1) Q_proj = question @ Wq1[:, :H]^T   (4096 x 75, padded to 80; TNC=5)
  {
    const int tm = (TQ * BB) / 16, waves = tm * 1;
    ptrnet_gemm_wmma<5><<<(waves + wpb - 1) / wpb, blk, 0, stream>>>(
        question, HH, Wq1, 2 * HH, Qproj, ATTP, ATT, HH, tm, 1);
  }
  // 2) P_proj = passage @ Wp1[:, :H]^T    (32768 x 75) — state-independent, reused twice
  {
    const int tm = (TP * BB) / 16, waves = tm * 1;
    ptrnet_gemm_wmma<5><<<(waves + wpb - 1) / wpb, blk, 0, stream>>>(
        passage, HH, Wp1, 2 * HH, Pproj, ATTP, ATT, HH, tm, 1);
  }
  // 3) Question pooling -> state (B,H)
  ptrnet_qpool<<<1, blk, 0, stream>>>(Qproj, question, qmask, Vq, Wq1, wq2, state);
  // 4) s_proj1 = state @ Wp1[:, H:]^T     (64 x 75)
  {
    const int tm = BB / 16, waves = tm * 1;
    ptrnet_gemm_wmma<5><<<(waves + wpb - 1) / wpb, blk, 0, stream>>>(
        state, HH, Wp1 + HH, 2 * HH, sproj1, ATTP, ATT, HH, tm, 1);
  }
  // 5) Attention pass 1 -> start_logits + cell_input
  ptrnet_pattn<<<BB, blk, 0, stream>>>(Pproj, passage, pmask, sproj1, wp2,
                                       out, cellin, 1);
  // 6) GRU GEMMs: gi = cellin @ W_ih^T, gh = state @ W_hh^T   (64 x 2304; TNC=4)
  {
    const int tm = BB / 16, chunks = (G3H / 16) / 4, waves = tm * chunks;
    ptrnet_gemm_wmma<4><<<(waves + wpb - 1) / wpb, blk, 0, stream>>>(
        cellin, HH, W_ih, HH, gi, G3H, G3H, HH, tm, chunks);
    ptrnet_gemm_wmma<4><<<(waves + wpb - 1) / wpb, blk, 0, stream>>>(
        state, HH, W_hh, HH, gh, G3H, G3H, HH, tm, chunks);
  }
  // 7) GRU combine -> state2
  ptrnet_gru<<<(BB * HH + 255) / 256, blk, 0, stream>>>(gi, gh, b_ih, b_hh,
                                                        state, state2);
  // 8) s_proj2 = state2 @ Wp1[:, H:]^T
  {
    const int tm = BB / 16, waves = tm * 1;
    ptrnet_gemm_wmma<5><<<(waves + wpb - 1) / wpb, blk, 0, stream>>>(
        state2, HH, Wp1 + HH, 2 * HH, sproj2, ATTP, ATT, HH, tm, 1);
  }
  // 9) Attention pass 2 -> end_logits (no cell_input needed)
  ptrnet_pattn<<<BB, blk, 0, stream>>>(Pproj, passage, pmask, sproj2, wp2,
                                       out + (size_t)BB * TP, nullptr, 0);
}
